// CorefGRU_60481729462793
// MI455X (gfx1250) — compile-verified, matrix operand order checked by reading, not run on hardware
//
#include <hip/hip_runtime.h>
#include <hip/hip_bf16.h>
#include <stdint.h>

// ---------------- CDNA5 WMMA types ----------------
typedef __bf16 v16bf __attribute__((ext_vector_type(16)));
typedef float  v8f   __attribute__((ext_vector_type(8)));

union BF16x16 { v16bf v; uint4 q[2]; };   // 32 bytes = 8 VGPRs per lane

// Problem constants (from reference setup)
#define BB   64
#define LL   512
#define DD   256     // Din == Dout
#define NTOK (BB*LL) // 32768
#define N3   768     // Wr|Wz|Wh concatenated

__device__ __forceinline__ unsigned short f2bf(float f) {
    unsigned u = __float_as_uint(f);
    u += 0x7FFFu + ((u >> 16) & 1u);       // round-to-nearest-even
    return (unsigned short)(u >> 16);
}
__device__ __forceinline__ float bf2f(unsigned short h) {
    return __uint_as_float(((unsigned)h) << 16);
}
__device__ __forceinline__ float sigmoidf_fast(float x) {
    return 1.0f / (1.0f + __expf(-x));
}
__device__ __forceinline__ float tanhf_fast(float x) {
    float t = __expf(-2.0f * fabsf(x));
    float r = (1.0f - t) / (1.0f + t);
    return copysignf(r, x);
}

// ---- A-tile loader: 16-bit A 16x32, layout per ISA 7.12.2 ----
// lane l: m = l&15 ; K runs = kcbase + 8*(l>>4) + {0..7}  (VGPR0-3)
//                     and     kcbase + 16 + 8*(l>>4) + {0..7} (VGPR4-7)
// With K-contiguous storage both runs are 16B-aligned b128 loads.
template <typename T>
__device__ __forceinline__ v16bf load_tile16(const T* rowbase /* &A[m][k0] */) {
    BF16x16 a;
    const uint4* p = (const uint4*)rowbase;
    a.q[0] = p[0];     // K = k0   .. k0+7
    a.q[1] = p[2];     // K = k0+16.. k0+23
    return a.v;
}

// =====================================================================
// Kernel 1: fp32 -> bf16 convert of the whole input (A-matrix feedstock)
// =====================================================================
__global__ __launch_bounds__(256) void k_convert_x(const float* __restrict__ x,
                                                   unsigned short* __restrict__ xb) {
    size_t i = (size_t)blockIdx.x * 256 + threadIdx.x;      // one float4 each
    const float4 v = ((const float4*)x)[i];
    uint2 o;
    o.x = (unsigned)f2bf(v.x) | ((unsigned)f2bf(v.y) << 16);
    o.y = (unsigned)f2bf(v.z) | ((unsigned)f2bf(v.w) << 16);
    ((uint2*)xb)[i] = o;
}

// =====================================================================
// Kernel 2: weight prep. Transpose W*/U* ([k][n] fp32) into K-contiguous
// bf16 B-matrix storage Wt[n_global][k], Ut[mat][n][k]; pack bias[768].
// =====================================================================
__global__ __launch_bounds__(256) void k_prep_w(
        const float* __restrict__ Wr, const float* __restrict__ Wz, const float* __restrict__ Wh,
        const float* __restrict__ Ur, const float* __restrict__ Uz, const float* __restrict__ Uh,
        const float* __restrict__ br, const float* __restrict__ bz, const float* __restrict__ bh,
        unsigned short* __restrict__ wt, unsigned short* __restrict__ ut,
        float* __restrict__ bias) {
    int idx = blockIdx.x * 256 + threadIdx.x;               // 3*256*256 threads
    int mat = idx >> 16;
    int n   = (idx >> 8) & 255;
    int k   = idx & 255;
    const float* W = (mat == 0) ? Wr : (mat == 1) ? Wz : Wh;
    const float* U = (mat == 0) ? Ur : (mat == 1) ? Uz : Uh;
    wt[(size_t)(mat * 256 + n) * 256 + k] = f2bf(W[k * 256 + n]);
    ut[(size_t)(mat * 256 + n) * 256 + k] = f2bf(U[k * 256 + n]);
    if (idx < N3) {
        bias[idx] = (idx < 256) ? br[idx] : (idx < 512) ? bz[idx - 256] : bh[idx - 512];
    }
}

// =====================================================================
// Kernel 3: attention weight  a = sigmoid(x . (k1 - k2))  (softmax-of-2)
// One wave32 per token; lane-parallel dot + xor-shuffle reduction.
// =====================================================================
__global__ __launch_bounds__(256) void k_alpha(const float* __restrict__ x,
                                               const float* __restrict__ k1,
                                               const float* __restrict__ k2,
                                               float* __restrict__ alpha) {
    int lane = threadIdx.x & 31;
    int tok  = blockIdx.x * 8 + (threadIdx.x >> 5);
    const float* xr = x + (size_t)tok * DD;
    float s = 0.f;
#pragma unroll
    for (int j = 0; j < 8; ++j) {
        int i = lane + 32 * j;
        s += xr[i] * (k1[i] - k2[i]);
    }
#pragma unroll
    for (int off = 16; off > 0; off >>= 1) s += __shfl_xor(s, off, 32);
    if (lane == 0) alpha[tok] = sigmoidf_fast(s);
}

// =====================================================================
// Kernel 4: big projection GEMM  xproj[32768,768] = xb @ Wt^T + bias
// 8 waves/block; wave = 16 rows x 64 cols (4 n-tiles), K-loop of 8
// v_wmma_f32_16x16x32_bf16. A reused across the 4 n-tiles.
// =====================================================================
__global__ __launch_bounds__(256) void k_xproj(const unsigned short* __restrict__ xb,
                                               const unsigned short* __restrict__ wt,
                                               const float* __restrict__ bias,
                                               float* __restrict__ xproj) {
    const int lane = threadIdx.x & 31, wave = threadIdx.x >> 5;
    const int hgrp = lane >> 4, lm = lane & 15;
    const int row0 = blockIdx.x * 128 + wave * 16;
    const int col0 = blockIdx.y * 64;
    v8f acc[4] = {};
#pragma unroll
    for (int kc = 0; kc < 8; ++kc) {
        const int k0 = kc * 32 + hgrp * 8;
        v16bf A = load_tile16(xb + (size_t)(row0 + lm) * DD + k0);
#pragma unroll
        for (int nt = 0; nt < 4; ++nt) {
            v16bf Bm = load_tile16(wt + (size_t)(col0 + nt * 16 + lm) * DD + k0);
            acc[nt] = __builtin_amdgcn_wmma_f32_16x16x32_bf16(
                          false, A, false, Bm, (short)0, acc[nt], false, false);
        }
    }
#pragma unroll
    for (int nt = 0; nt < 4; ++nt) {
#pragma unroll
        for (int r = 0; r < 8; ++r) {
            int m = hgrp * 8 + r;                 // C layout: lanes16-31 hold M=r+8
            int n = col0 + nt * 16 + lm;
            xproj[(size_t)(row0 + m) * N3 + n] = acc[nt][r] + bias[n];
        }
    }
}

// =====================================================================
// Kernel 5: the sequential recurrence. ONE persistent workgroup,
// 1024 threads = 32 waves, 512 steps with workgroup barriers.
//  - history lives in d_out ([B,L,D] == ys), coref gathers read it back
//  - mt staged in LDS as bf16 (A-matrix), U weights streamed from L2
//  - wave w owns output tile positions 2w,2w+1 for ALL THREE U matrices
//    so gate math stays in registers.
// =====================================================================
__global__ __launch_bounds__(1024) void k_recur(const float* __restrict__ alpha,
                                                const int* __restrict__ ci,
                                                const float* __restrict__ xproj,
                                                const unsigned short* __restrict__ ut,
                                                float* __restrict__ out) {
    __shared__ __align__(16) unsigned short mt[BB][DD + 8];   // +8 pad keeps 16B align, skews banks

    const int tid  = threadIdx.x;
    const int lane = tid & 31, wave = tid >> 5;
    const int p0   = wave * 2;          // two tile positions per wave (same 16 rows)
    const int pr   = p0 >> 4;           // tile row   (b block, 0..3)
    const int pc0  = p0 & 15;           // tile col 0 (d block)
    const int hgrp = lane >> 4, lm = lane & 15;

    // Warm L2 with the recurrent weights (384 KB, resident all 512 steps).
    for (int i = tid * 64; i < 3 * 256 * 256; i += 1024 * 64)
        __builtin_prefetch(ut + i, 0, 1);

    const int mb = tid >> 4;            // mt builder: 16 threads per batch row
    const int md0 = (tid & 15) * 16;

    for (int t = 0; t < LL; ++t) {
        // ---- phase A: build mt = [a*h_last[:,:128] | (1-a)*coref[:,128:]] ----
        {
            const float a = alpha[mb * LL + t];
            if (md0 < 128) {
                if (t > 0) {
                    const float* hp = out + ((size_t)mb * LL + (t - 1)) * DD;
#pragma unroll
                    for (int d = md0; d < md0 + 16; ++d) mt[mb][d] = f2bf(a * hp[d]);
                } else {
#pragma unroll
                    for (int d = md0; d < md0 + 16; ++d) mt[mb][d] = 0;
                }
            } else {
                const int c = ci[mb * LL + t];
                if (c > 0) {
                    const float* cp = out + ((size_t)mb * LL + (c - 1)) * DD;
#pragma unroll
                    for (int d = md0; d < md0 + 16; ++d) mt[mb][d] = f2bf((1.f - a) * cp[d]);
                } else {
#pragma unroll
                    for (int d = md0; d < md0 + 16; ++d) mt[mb][d] = 0;
                }
            }
        }
        __syncthreads();

        // ---- phase B: acc[pi][m] = (mt @ U_m) tiles, f32 accumulate ----
        v8f acc[2][3] = {};
#pragma unroll
        for (int kc = 0; kc < 8; ++kc) {
            const int k0 = kc * 32 + hgrp * 8;
            v16bf A = load_tile16(&mt[pr * 16 + lm][k0]);      // ds_load_b128 x2
#pragma unroll
            for (int pi = 0; pi < 2; ++pi) {
                const int n = (pc0 + pi) * 16 + lm;
#pragma unroll
                for (int m = 0; m < 3; ++m) {
                    v16bf Bv = load_tile16(ut + ((size_t)m * 256 + n) * DD + k0);
                    acc[pi][m] = __builtin_amdgcn_wmma_f32_16x16x32_bf16(
                                     false, A, false, Bv, (short)0, acc[pi][m], false, false);
                }
            }
        }

        // ---- phase C: gates + blend, write h to d_out (== history) ----
#pragma unroll
        for (int pi = 0; pi < 2; ++pi) {
            const int d = (pc0 + pi) * 16 + lm;
#pragma unroll
            for (int r = 0; r < 8; ++r) {
                const int b   = pr * 16 + hgrp * 8 + r;
                const size_t tok = (size_t)b * LL + t;
                const float* xp = xproj + tok * N3 + d;
                const float xr = xp[0], xz = xp[256], xh = xp[512];
                const float mtv = bf2f(mt[b][d]);
                const float rg = sigmoidf_fast(xr + acc[pi][0][r]);
                const float zg = sigmoidf_fast(xz + acc[pi][1][r]);
                const float hc = tanhf_fast(xh + rg * acc[pi][2][r]);
                out[tok * DD + d] = (1.f - zg) * mtv + zg * hc;
            }
        }
        __threadfence();     // make history stores visible before next-step gathers
        __syncthreads();
    }
}

// =====================================================================
// Launcher
// =====================================================================
extern "C" void kernel_launch(void* const* d_in, const int* in_sizes, int n_in,
                              void* d_out, int out_size, void* d_ws, size_t ws_size,
                              hipStream_t stream) {
    const float* inp = (const float*)d_in[0];
    const int*   ci  = (const int*)d_in[1];
    const float* Wr  = (const float*)d_in[2];
    const float* br  = (const float*)d_in[3];
    const float* Ur  = (const float*)d_in[4];
    const float* Wz  = (const float*)d_in[5];
    const float* bz  = (const float*)d_in[6];
    const float* Uz  = (const float*)d_in[7];
    const float* Wh  = (const float*)d_in[8];
    const float* bh  = (const float*)d_in[9];
    const float* Uh  = (const float*)d_in[10];
    const float* k1  = (const float*)d_in[11];
    const float* k2  = (const float*)d_in[12];
    float* out = (float*)d_out;

    // Workspace carve-up (~113 MB total, all 256B aligned)
    char* ws = (char*)d_ws;
    const size_t XB_BYTES    = (size_t)NTOK * DD * 2;         // 16 MB  bf16 input
    const size_t WT_BYTES    = (size_t)N3 * DD * 2;           // 384 KB bf16 Wt
    const size_t UT_BYTES    = (size_t)3 * DD * DD * 2;       // 384 KB bf16 Ut
    const size_t BIAS_BYTES  = (size_t)N3 * 4;                // 3 KB
    const size_t ALPHA_BYTES = (size_t)NTOK * 4;              // 128 KB
    unsigned short* xb    = (unsigned short*)(ws);
    unsigned short* wt    = (unsigned short*)(ws + XB_BYTES);
    unsigned short* ut    = (unsigned short*)(ws + XB_BYTES + WT_BYTES);
    float*          bias  = (float*)(ws + XB_BYTES + WT_BYTES + UT_BYTES);
    float*          alpha = (float*)(ws + XB_BYTES + WT_BYTES + UT_BYTES + BIAS_BYTES);
    float*          xproj = (float*)(ws + XB_BYTES + WT_BYTES + UT_BYTES + BIAS_BYTES + ALPHA_BYTES);

    // 1) bf16 convert of input: 32768*256/4 float4s
    k_convert_x<<<(NTOK * DD / 4) / 256, 256, 0, stream>>>(inp, xb);
    // 2) transpose+convert all weights, pack bias
    k_prep_w<<<(3 * 256 * 256) / 256, 256, 0, stream>>>(Wr, Wz, Wh, Ur, Uz, Uh,
                                                        br, bz, bh, wt, ut, bias);
    // 3) a = sigmoid(x.(k1-k2)) for every token
    k_alpha<<<NTOK / 8, 256, 0, stream>>>(inp, k1, k2, alpha);
    // 4) xproj = x @ [Wr|Wz|Wh] + bias   (token-parallel WMMA GEMM)
    k_xproj<<<dim3(NTOK / 128, N3 / 64), 256, 0, stream>>>(xb, wt, bias, xproj);
    // 5) sequential GRU recurrence, single persistent workgroup
    k_recur<<<1, 1024, 0, stream>>>(alpha, ci, xproj, ut, out);
}
